// META3_74569222193915
// MI455X (gfx1250) — compile-verified, matrix operand order checked by reading, not run on hardware
//
#include <hip/hip_runtime.h>
#include <hip/hip_bf16.h>

typedef __attribute__((ext_vector_type(16))) _Float16 v16h;
typedef __attribute__((ext_vector_type(8)))  float    v8f;

#define HID 50
#define D3  15

__device__ __forceinline__ v8f wmma_f16(v16h a, v16h b, v8f c) {
  // v_wmma_f32_16x16x32_f16: D = A(16x32 f16) * B(32x16 f16) + C(16x16 f32)
  return __builtin_amdgcn_wmma_f32_16x16x32_f16(
      /*neg_a=*/false, a, /*neg_b=*/false, b,
      /*c_mod=*/(short)0, c, /*reuse_a=*/false, /*reuse_b=*/false);
}

// ---------------------------------------------------------------------------
// Kernel 0: zero workspace accumulators
// ---------------------------------------------------------------------------
__global__ void zero_kernel(float* __restrict__ p, size_t n) {
  size_t i = (size_t)blockIdx.x * blockDim.x + threadIdx.x;
  size_t stride = (size_t)gridDim.x * blockDim.x;
  for (; i < n; i += stride) p[i] = 0.0f;
}

// ---------------------------------------------------------------------------
// Kernel 1: fused edge-MLP + node-MLP over 16-edge tiles, WMMA f16 path.
// One wave32 per tile; grid-stride over tiles. Single-wave workgroups so
// __syncthreads is just a fence (barrier -> S_NOP on 1-wave WGs per ISA).
//
// Fragment layout convention (ISA 7.12.2, wave32):
//   A (16x32 f16): lane = half*16 + m ; element j -> K = (j>>3)*16 + half*8 + (j&7)
//   B (32x16 f16): lane = half*16 + n ; same K mapping
//   C/D (16x16 f32): vgpr r -> M = r + 8*half ; N = lane & 15
// ---------------------------------------------------------------------------
__global__ __launch_bounds__(32) void edge_node_kernel(
    const float* __restrict__ x, const int* __restrict__ ei,
    const float* __restrict__ ea,
    const float* __restrict__ ew1, const float* __restrict__ eb1,
    const float* __restrict__ ew2, const float* __restrict__ eb2,
    const float* __restrict__ nw1, const float* __restrict__ nb1,
    const float* __restrict__ nw2, const float* __restrict__ nb2,
    float* __restrict__ x2sum, float* __restrict__ cnt,
    int E, int numTiles)
{
  __shared__ __align__(32) _Float16 wE1[4][32][16];  // edge L1: 4 N-tiles of 16
  __shared__ __align__(32) _Float16 wE2[2][32][16];  // edge L2: 2 K-chunks of 32
  __shared__ __align__(32) _Float16 wN1[4][32][16];  // node L1
  __shared__ __align__(32) _Float16 wN2[2][32][16];  // node L2
  __shared__ __align__(32) _Float16 hA[32][32];      // hidden acts in A-frag layout (K=0..63)
  __shared__ __align__(32) _Float16 nA[32][16];      // node-MLP input in A-frag layout (K=0..31)
  __shared__ float bE1s[64], bN1s[64], bE2s[16], bN2s[16];
  __shared__ int   rowS[16];

  const int lane = threadIdx.x;       // 0..31
  const int half = lane >> 4;         // 0 or 1
  const int n15  = lane & 15;

  // ---- pack weights into B-fragment layout in LDS (once per block) ----
  #pragma unroll
  for (int t = 0; t < 4; ++t) {
    #pragma unroll
    for (int j = 0; j < 16; ++j) {
      int K = ((j >> 3) << 4) + (half << 3) + (j & 7);
      int n = 16 * t + n15;
      wE1[t][lane][j] = (K < 9  && n < HID) ? (_Float16)ew1[K * HID + n] : (_Float16)0.0f;
      wN1[t][lane][j] = (K < 18 && n < HID) ? (_Float16)nw1[K * HID + n] : (_Float16)0.0f;
    }
  }
  #pragma unroll
  for (int c = 0; c < 2; ++c) {
    #pragma unroll
    for (int j = 0; j < 16; ++j) {
      int K = 32 * c + ((j >> 3) << 4) + (half << 3) + (j & 7);
      wE2[c][lane][j] = (K < HID && n15 < D3) ? (_Float16)ew2[K * D3 + n15] : (_Float16)0.0f;
      wN2[c][lane][j] = (K < HID && n15 < D3) ? (_Float16)nw2[K * D3 + n15] : (_Float16)0.0f;
    }
  }
  for (int i = lane; i < 64; i += 32) {
    bE1s[i] = (i < HID) ? eb1[i] : 0.0f;
    bN1s[i] = (i < HID) ? nb1[i] : 0.0f;
  }
  if (lane < 16) {
    bE2s[lane] = (lane < D3) ? eb2[lane] : 0.0f;
    bN2s[lane] = (lane < D3) ? nb2[lane] : 0.0f;
  }
  {  // zero nA once: K=18..31 pad region stays zero forever
    _Float16* p = &nA[0][0];
    for (int i = lane; i < 32 * 16; i += 32) p[i] = (_Float16)0.0f;
  }
  __syncthreads();

  for (int tile = blockIdx.x; tile < numTiles; tile += gridDim.x) {
    const int e = tile * 16 + n15;
    const bool valid = (e < E);
    int row = 0, col = 0;
    float xr0=0,xr1=0,xr2=0, xc0=0,xc1=0,xc2=0, ea0=0,ea1=0,ea2=0;
    if (valid) {
      row = ei[e]; col = ei[E + e];
      xr0 = x[row*3+0]; xr1 = x[row*3+1]; xr2 = x[row*3+2];
      xc0 = x[col*3+0]; xc1 = x[col*3+1]; xc2 = x[col*3+2];
      ea0 = ea[(size_t)e*3+0]; ea1 = ea[(size_t)e*3+1]; ea2 = ea[(size_t)e*3+2];
    }
    if (half == 0) rowS[n15] = valid ? row : -1;

    {   // prefetch next tile's streamed data into L2 (global_prefetch_b8)
      int tn = tile + gridDim.x;
      if (tn < numTiles) {
        __builtin_prefetch(&ea[(size_t)(tn * 16 + n15) * 3], 0, 1);
        __builtin_prefetch(&ei[tn * 16 + n15], 0, 1);
      }
    }

    // ---- edge MLP layer 1: A = [x[row], x[col], edge_attr] (K pad 9->32) ----
    v16h a = {};
    if (half == 0) {       // lanes 0-15 hold K=0..7 (j=0..7)
      a[0]=(_Float16)xr0; a[1]=(_Float16)xr1; a[2]=(_Float16)xr2;
      a[3]=(_Float16)xc0; a[4]=(_Float16)xc1; a[5]=(_Float16)xc2;
      a[6]=(_Float16)ea0; a[7]=(_Float16)ea1;
    } else {               // lanes 16-31 hold K=8..15; only K=8 valid
      a[0]=(_Float16)ea2;
    }
    v8f acc[4] = {};
    #pragma unroll
    for (int t = 0; t < 4; ++t) {
      v16h b = *(const v16h*)&wE1[t][lane][0];
      acc[t] = wmma_f16(a, b, acc[t]);
    }
    // bias + ReLU, restage hidden (16x64) into A-fragment layout in hA
    #pragma unroll
    for (int t = 0; t < 4; ++t) {
      float bias = bE1s[16 * t + n15];
      #pragma unroll
      for (int r = 0; r < 8; ++r) {
        int mc = r + 8 * half;
        int kcol = 16 * t + n15;
        float v = fmaxf(acc[t][r] + bias, 0.0f);   // pad cols (>=50) -> 0
        hA[(((kcol >> 3) & 1) << 4) | mc][((kcol >> 4) << 3) | (kcol & 7)] = (_Float16)v;
      }
    }
    __syncthreads();

    // ---- edge MLP layer 2: K = 50 padded to 64 -> two k=32 WMMAs ----
    v16h a0 = *(const v16h*)&hA[lane][0];
    v16h a1 = *(const v16h*)&hA[lane][16];
    v8f e2c = {};
    e2c = wmma_f16(a0, *(const v16h*)&wE2[0][lane][0], e2c);
    e2c = wmma_f16(a1, *(const v16h*)&wE2[1][lane][0], e2c);
    __syncthreads();   // everyone done reading hA

    // ---- stage node-MLP input [x[col](3), e2(15)] into nA (K pad 18->32) ----
    if (half == 0) {   // lane == m
      nA[n15][0] = (_Float16)xc0;
      nA[n15][1] = (_Float16)xc1;
      nA[n15][2] = (_Float16)xc2;
    }
    if (n15 < D3) {
      float bias2 = bE2s[n15];
      #pragma unroll
      for (int r = 0; r < 8; ++r) {
        int mc = r + 8 * half;
        int Kf = 3 + n15;
        float v = e2c[r] + bias2;                  // edge MLP L2 is linear
        nA[(((Kf >> 3) & 1) << 4) | mc][((Kf >> 4) << 3) | (Kf & 7)] = (_Float16)v;
      }
    }
    __syncthreads();

    // ---- node MLP layer 1 ----
    v16h an = *(const v16h*)&nA[lane][0];
    v8f nacc[4] = {};
    #pragma unroll
    for (int t = 0; t < 4; ++t) {
      v16h b = *(const v16h*)&wN1[t][lane][0];
      nacc[t] = wmma_f16(an, b, nacc[t]);
    }
    #pragma unroll
    for (int t = 0; t < 4; ++t) {
      float bias = bN1s[16 * t + n15];
      #pragma unroll
      for (int r = 0; r < 8; ++r) {
        int mc = r + 8 * half;
        int kcol = 16 * t + n15;
        float v = fmaxf(nacc[t][r] + bias, 0.0f);
        hA[(((kcol >> 3) & 1) << 4) | mc][((kcol >> 4) << 3) | (kcol & 7)] = (_Float16)v;
      }
    }
    __syncthreads();

    // ---- node MLP layer 2 ----
    v16h c0 = *(const v16h*)&hA[lane][0];
    v16h c1 = *(const v16h*)&hA[lane][16];
    v8f hn = {};
    hn = wmma_f16(c0, *(const v16h*)&wN2[0][lane][0], hn);
    hn = wmma_f16(c1, *(const v16h*)&wN2[1][lane][0], hn);

    // ---- scatter segment-sum over row (atomics resolve in L2) ----
    if (n15 < D3) {
      float bias2 = bN2s[n15];
      #pragma unroll
      for (int r = 0; r < 8; ++r) {
        int mc = r + 8 * half;
        int rw = rowS[mc];
        if (rw >= 0) atomicAdd(&x2sum[(size_t)rw * 16 + n15], hn[r] + bias2);
      }
    }
    if (n15 == 0) {
      #pragma unroll
      for (int r = 0; r < 8; ++r) {
        int mc = r + 8 * half;
        int rw = rowS[mc];
        if (rw >= 0) atomicAdd(&cnt[rw], 1.0f);
      }
    }
    __syncthreads();   // protect LDS stages + rowS across iterations
  }
}

// ---------------------------------------------------------------------------
// Kernel 2: per-node finalize x2 = sum/cnt ; accumulate per-graph sums
//   gsum  += x2            (input to global MLP segment mean)
//   zsum  += relu(x2)      (readout segment mean, node part)
//   gcnt  += 1
// ---------------------------------------------------------------------------
__global__ __launch_bounds__(256) void node_kernel(
    const float* __restrict__ x2sum, const float* __restrict__ cnt,
    const int* __restrict__ batch,
    float* __restrict__ gsum, float* __restrict__ gcnt,
    float* __restrict__ zsum, int N)
{
  int i = blockIdx.x * blockDim.x + threadIdx.x;
  if (i >= N) return;
  float c   = cnt[i];
  float inv = 1.0f / fmaxf(c, 1.0f);
  int   b   = batch[i];
  #pragma unroll
  for (int k = 0; k < D3; ++k) {
    float v = x2sum[(size_t)i * 16 + k] * inv;
    atomicAdd(&gsum[(size_t)b * 16 + k], v);
    atomicAdd(&zsum[(size_t)b * 16 + k], fmaxf(v, 0.0f));
  }
  atomicAdd(&gcnt[b], 1.0f);
}

// ---------------------------------------------------------------------------
// Kernel 3: global MLP + readout + fc1/BN/relu + fc2 + log_softmax (B=256)
// ---------------------------------------------------------------------------
__global__ __launch_bounds__(256) void head_kernel(
    const float* __restrict__ u,
    const float* __restrict__ gw1, const float* __restrict__ gb1,
    const float* __restrict__ gw2, const float* __restrict__ gb2,
    const float* __restrict__ fc1w, const float* __restrict__ fc1b,
    const float* __restrict__ bng, const float* __restrict__ bnb,
    const float* __restrict__ fc2w, const float* __restrict__ fc2b,
    const float* __restrict__ gsum, const float* __restrict__ gcnt,
    const float* __restrict__ zsum,
    float* __restrict__ out, int B)
{
  int b = blockIdx.x * blockDim.x + threadIdx.x;
  if (b >= B) return;
  float gc  = gcnt[b];
  float inv = 1.0f / fmaxf(gc, 1.0f);

  float in0[16];
  in0[0] = u[b];
  #pragma unroll
  for (int k = 0; k < D3; ++k) in0[k + 1] = gsum[(size_t)b * 16 + k] * inv;

  float u2[D3];
  #pragma unroll
  for (int k = 0; k < D3; ++k) u2[k] = gb2[k];
  for (int h = 0; h < HID; ++h) {
    float s = gb1[h];
    #pragma unroll
    for (int i = 0; i < 16; ++i) s += in0[i] * gw1[i * HID + h];
    s = fmaxf(s, 0.0f);
    #pragma unroll
    for (int k = 0; k < D3; ++k) u2[k] += s * gw2[h * D3 + k];
  }

  // readout segment mean: nodes of graph b plus its u2 row (count = gc + 1)
  float zinv = 1.0f / (gc + 1.0f);
  float g[D3];
  #pragma unroll
  for (int k = 0; k < D3; ++k)
    g[k] = (zsum[(size_t)b * 16 + k] + fmaxf(u2[k], 0.0f)) * zinv;

  float h1[10];
  const float bscale = rsqrtf(1.0f + 1e-5f);
  #pragma unroll
  for (int d = 0; d < 10; ++d) {
    float s = fc1b[d];
    #pragma unroll
    for (int k = 0; k < D3; ++k) s += g[k] * fc1w[k * 10 + d];
    s = s * (bng[d] * bscale) + bnb[d];     // BN eval (mean=0, var=1)
    h1[d] = fmaxf(s, 0.0f);
  }

  float lg[6], mx = -1e30f;
  #pragma unroll
  for (int j = 0; j < 6; ++j) {
    float s = fc2b[j];
    #pragma unroll
    for (int d = 0; d < 10; ++d) s += h1[d] * fc2w[d * 6 + j];
    lg[j] = s; mx = fmaxf(mx, s);
  }
  float se = 0.0f;
  #pragma unroll
  for (int j = 0; j < 6; ++j) se += __expf(lg[j] - mx);
  float lse = __logf(se);
  #pragma unroll
  for (int j = 0; j < 6; ++j) out[b * 6 + j] = lg[j] - mx - lse;
}

// ---------------------------------------------------------------------------
extern "C" void kernel_launch(void* const* d_in, const int* in_sizes, int n_in,
                              void* d_out, int out_size, void* d_ws, size_t ws_size,
                              hipStream_t stream) {
  const float* x     = (const float*)d_in[0];
  const int*   ei    = (const int*)  d_in[1];
  const float* ea    = (const float*)d_in[2];
  const float* u     = (const float*)d_in[3];
  const int*   batch = (const int*)  d_in[4];
  const float* ew1 = (const float*)d_in[5];
  const float* eb1 = (const float*)d_in[6];
  const float* ew2 = (const float*)d_in[7];
  const float* eb2 = (const float*)d_in[8];
  const float* nw1 = (const float*)d_in[9];
  const float* nb1 = (const float*)d_in[10];
  const float* nw2 = (const float*)d_in[11];
  const float* nb2 = (const float*)d_in[12];
  const float* gw1 = (const float*)d_in[13];
  const float* gb1 = (const float*)d_in[14];
  const float* gw2 = (const float*)d_in[15];
  const float* gb2 = (const float*)d_in[16];
  const float* fc1w = (const float*)d_in[17];
  const float* fc1b = (const float*)d_in[18];
  const float* bng  = (const float*)d_in[19];
  const float* bnb  = (const float*)d_in[20];
  const float* fc2w = (const float*)d_in[21];
  const float* fc2b = (const float*)d_in[22];

  const int N = in_sizes[0] / 3;
  const int E = in_sizes[1] / 2;
  const int B = in_sizes[3];

  float* ws    = (float*)d_ws;
  float* x2sum = ws;                          // N*16
  float* cnt   = x2sum + (size_t)N * 16;      // N
  float* gsum  = cnt   + N;                   // B*16
  float* gcnt  = gsum  + (size_t)B * 16;      // B
  float* zsum  = gcnt  + B;                   // B*16
  const size_t totalF = (size_t)N * 16 + N + (size_t)B * 16 + B + (size_t)B * 16;

  zero_kernel<<<1024, 256, 0, stream>>>(ws, totalF);

  const int numTiles = (E + 15) / 16;
  const int blocks   = numTiles < 4096 ? numTiles : 4096;
  edge_node_kernel<<<blocks, 32, 0, stream>>>(
      x, ei, ea, ew1, eb1, ew2, eb2, nw1, nb1, nw2, nb2,
      x2sum, cnt, E, numTiles);

  node_kernel<<<(N + 255) / 256, 256, 0, stream>>>(
      x2sum, cnt, batch, gsum, gcnt, zsum, N);

  head_kernel<<<(B + 255) / 256, 256, 0, stream>>>(
      u, gw1, gb1, gw2, gb2, fc1w, fc1b, bng, bnb, fc2w, fc2b,
      gsum, gcnt, zsum, (float*)d_out, B);
}
